// MoeLayer_89464168776159
// MI455X (gfx1250) — matrix operand order repaired and checked by reading, not compiled, not run on hardware
//
#include <hip/hip_runtime.h>
#include <hip/hip_bf16.h>
#include <math.h>

// ---------------------------------------------------------------------------
// MoE layer (top-2 of 8 experts, SwiGLU) for MI455X (gfx1250, wave32, WMMA).
//
// Routed FLOPs = T*K*3*2*D*F ~= 103 GFLOP; weights 201 MB f32 (~9us @23.3TB/s)
// -> compute-bound on the matrix units. All three projections run through
// v_wmma_f32_16x16x32_bf16 on gathered (routed-only) token lists, with
// double-buffered LDS tiles filled by GLOBAL_LOAD_ASYNC_TO_LDS_B128
// (ASYNCcnt-tracked direct global->LDS copies, no VGPR round trip).
// ---------------------------------------------------------------------------

#define T_TOK 4096   // B*S
#define DMODEL 1024
#define NEXP 8
#define FDIM 2048

typedef __attribute__((ext_vector_type(16))) __bf16          v16bf;
typedef __attribute__((ext_vector_type(8)))  float           v8f;
typedef __attribute__((ext_vector_type(8)))  unsigned short  v8u16;
typedef __attribute__((ext_vector_type(16))) unsigned short  v16u16;

__device__ __forceinline__ unsigned short f2bf(float f) {
  // round-to-nearest-even f32 -> bf16
  unsigned int u = __builtin_bit_cast(unsigned int, f);
  unsigned int r = u + 0x7FFFu + ((u >> 16) & 1u);
  return (unsigned short)(r >> 16);
}

// Async global->LDS 16B copy (gfx1250). VDST = 32-bit LDS byte offset (low 32
// bits of the generic pointer to a __shared__ object), VADDR = 64-bit global
// address. Tracked by ASYNCcnt; in-order completion among async loads.
__device__ __forceinline__ void async_cp16(void* lds_dst, const void* gsrc) {
  unsigned loff = (unsigned)(uintptr_t)lds_dst;  // LDS offset = low 32 bits
  asm volatile("global_load_async_to_lds_b128 %0, %1, off"
               :: "v"(loff), "v"(gsrc)
               : "memory");
}
__device__ __forceinline__ void wait_async_le(int n) {
  if (n == 0)      asm volatile("s_wait_asynccnt 0x0" ::: "memory");
  else if (n == 3) asm volatile("s_wait_asynccnt 0x3" ::: "memory");
  else             asm volatile("s_wait_asynccnt 0x5" ::: "memory");
}

// ---------------------------------------------------------------------------
// Gate: logits, softmax stats for l_aux, top-2 routing lists (gathered).
// One wave32 per token; 8 tokens per 256-thread block.
// ---------------------------------------------------------------------------
__global__ __launch_bounds__(256) void gate_kernel(
    const float* __restrict__ x, const float* __restrict__ gatew,
    float* __restrict__ logits_out,  // [T, E] (tail of d_out)
    int* __restrict__ cnt, float* __restrict__ me_sum, float* __restrict__ ce_sum,
    int* __restrict__ idx, float* __restrict__ gwt) {
  int tok  = blockIdx.x * (blockDim.x >> 5) + (threadIdx.x >> 5);
  int lane = threadIdx.x & 31;
  if (tok >= T_TOK) return;

  float acc[NEXP];
#pragma unroll
  for (int e = 0; e < NEXP; ++e) acc[e] = 0.f;
  const float* xr = x + (size_t)tok * DMODEL;
  for (int d = lane; d < DMODEL; d += 32) {
    float xv = xr[d];
    const float* g = gatew + (size_t)d * NEXP;  // gate_w is [D, E] row-major
#pragma unroll
    for (int e = 0; e < NEXP; ++e) acc[e] = fmaf(xv, g[e], acc[e]);
  }
#pragma unroll
  for (int e = 0; e < NEXP; ++e)
    for (int off = 16; off; off >>= 1) acc[e] += __shfl_xor(acc[e], off, 32);

  if (lane == 0) {
    float m = acc[0]; int am = 0;
    for (int e = 1; e < NEXP; ++e) if (acc[e] > m) { m = acc[e]; am = e; }
    float s = 0.f, p[NEXP];
    for (int e = 0; e < NEXP; ++e) { p[e] = __expf(acc[e] - m); s += p[e]; }
    float inv = 1.f / s;
    for (int e = 0; e < NEXP; ++e) {
      atomicAdd(&me_sum[e], p[e] * inv);
      logits_out[(size_t)tok * NEXP + e] = acc[e];
    }
    atomicAdd(&ce_sum[am], 1.0f);

    // top-2 on LOGITS (matches reference); softmax over the two logits
    int e1 = am; float l1 = m;
    int e2 = 0;  float l2 = -INFINITY;
    for (int e = 0; e < NEXP; ++e)
      if (e != e1 && acc[e] > l2) { l2 = acc[e]; e2 = e; }
    float wa = 1.f / (1.f + __expf(l2 - l1));
    float wb = 1.f - wa;

    int p1 = atomicAdd(&cnt[e1], 1);
    idx[e1 * T_TOK + p1] = tok; gwt[e1 * T_TOK + p1] = wa;
    int p2 = atomicAdd(&cnt[e2], 1);
    idx[e2 * T_TOK + p2] = tok; gwt[e2 * T_TOK + p2] = wb;
  }
}

__global__ void laux_kernel(const float* me_sum, const float* ce_sum, float* out) {
  float s = 0.f;
  for (int e = 0; e < NEXP; ++e) s += me_sum[e] * ce_sum[e];
  // mean(me*ce)*E*E with me=me_sum/T, ce=ce_sum/T  ->  E * sum / T^2
  *out = (float)NEXP * s / ((float)T_TOK * (float)T_TOK);
}

// ---------------------------------------------------------------------------
// f32 -> bf16 converters (x straight; weights transposed to N-major so the
// WMMA B fragment is one contiguous 32B read per lane).
// ---------------------------------------------------------------------------
__global__ void cvt_x(const float* __restrict__ x, unsigned short* __restrict__ xb, int n) {
  int i = blockIdx.x * blockDim.x + threadIdx.x;
  if (i < n) xb[i] = f2bf(x[i]);
}

// in[e][R][C] f32 -> out[e][C][R] bf16   (R, C multiples of 32)
__global__ __launch_bounds__(256) void tconv(const float* __restrict__ in,
                                             unsigned short* __restrict__ out,
                                             int R, int C) {
  __shared__ float tile[32][33];
  int e = blockIdx.z;
  const float* ip = in + (size_t)e * R * C;
  unsigned short* op = out + (size_t)e * R * C;
  int c0 = blockIdx.x * 32, r0 = blockIdx.y * 32;
  int tx = threadIdx.x & 31, ty = threadIdx.x >> 5;  // 8 rows of 32
  for (int i = ty; i < 32; i += 8)
    tile[i][tx] = ip[(size_t)(r0 + i) * C + c0 + tx];
  __syncthreads();
  for (int i = ty; i < 32; i += 8)
    op[(size_t)(c0 + i) * R + r0 + tx] = f2bf(tile[tx][i]);
}

// ---------------------------------------------------------------------------
// WMMA fragment helpers (wave32; A and B use the same contiguous-K order).
// A (16x32 bf16): lane row = lane&15; 16 contiguous K at
//                 (lane>>4)*16.
// B (32x16 bf16, N-major storage): lane col = lane&15; 16 contiguous K at
//                 (lane>>4)*16.
// D (16x16 f32): elem v -> row (lane>>4)*8 + v, col lane&15.
// ---------------------------------------------------------------------------
__device__ __forceinline__ v16bf load_a_frag(const unsigned short* row_base, int hi) {
  v8u16 lo = *(const v8u16*)(row_base + hi * 16);
  v8u16 hi8 = *(const v8u16*)(row_base + hi * 16 + 8);
  v16u16 u;
#pragma unroll
  for (int i = 0; i < 8; ++i) { u[i] = lo[i]; u[i + 8] = hi8[i]; }
  return __builtin_bit_cast(v16bf, u);
}
__device__ __forceinline__ v16bf load_b_frag(const unsigned short* col_base, int hi) {
  v16u16 u = *(const v16u16*)(col_base + hi * 16);
  return __builtin_bit_cast(v16bf, u);
}

// ---------------------------------------------------------------------------
// GEMM1 + SwiGLU: H[m, :] = silu(Xg@W1) * (Xg@W3), gathered rows.
// Block = 256 thr (8 waves), tile 64(M) x 128(N); wave tile 32x32.
// Double-buffered LDS, filled with async global->LDS copies (5 per thread
// per k-step: 1xA + 2xW1 + 2xW3).
// ---------------------------------------------------------------------------
__global__ __launch_bounds__(256) void gemm1_swiglu(
    const unsigned short* __restrict__ xb,   // [T, D] bf16
    const unsigned short* __restrict__ w1b,  // [E, F, D] bf16 (N-major)
    const unsigned short* __restrict__ w3b,  // [E, F, D] bf16
    unsigned short* __restrict__ hb,         // [E, T, F] bf16
    const int* __restrict__ cnt, const int* __restrict__ idx) {
  int e = blockIdx.z;
  int n0 = blockIdx.x * 128;
  int m0 = blockIdx.y * 64;
  int ce = cnt[e];
  if (m0 >= ce) return;

  __shared__ __align__(32) unsigned short sA[2][64][32];
  __shared__ __align__(32) unsigned short sB1[2][128][32];
  __shared__ __align__(32) unsigned short sB3[2][128][32];
  __shared__ int sIdx[64];

  int tid = threadIdx.x, lane = tid & 31, wave = tid >> 5;
  int wm = wave >> 2, wn = wave & 3;
  int hi = lane >> 4, lo = lane & 15;

  if (tid < 64) {
    int r = m0 + tid;
    // clamp out-of-range gathered rows to token 0: finite garbage in tile
    // rows >= cnt, which only feed H rows that GEMM2's store guard discards.
    sIdx[tid] = (r < ce) ? idx[e * T_TOK + r] : 0;
  }
  __syncthreads();

  int ar = tid >> 2, ach = (tid & 3) * 8;     // A staging: 16B per thread
  int br = tid >> 1, bhf = (tid & 1) * 16;    // B staging: 32B per thread
  int ag = sIdx[ar];

  auto stage = [&](int buf, int k0) {
    async_cp16(&sA[buf][ar][ach], xb + (size_t)ag * DMODEL + k0 + ach);
    size_t base = ((size_t)e * FDIM + n0 + br) * DMODEL + k0 + bhf;
    async_cp16(&sB1[buf][br][bhf],     w1b + base);
    async_cp16(&sB1[buf][br][bhf + 8], w1b + base + 8);
    async_cp16(&sB3[buf][br][bhf],     w3b + base);
    async_cp16(&sB3[buf][br][bhf + 8], w3b + base + 8);
  };

  v8f accU[2][2] = {{{0.f}}}, accV[2][2] = {{{0.f}}};

  const int NKB = DMODEL / 32;
  stage(0, 0);
  for (int kb = 0; kb < NKB; ++kb) {
    int cur = kb & 1;
    if (kb + 1 < NKB) {
      stage(cur ^ 1, (kb + 1) * 32);   // prefetch next tile into other buffer
      wait_async_le(5);                // oldest batch (this tile) has landed
    } else {
      wait_async_le(0);
    }
    __syncthreads();                   // publish LDS tile block-wide

    v16bf af[2];
#pragma unroll
    for (int sm = 0; sm < 2; ++sm)
      af[sm] = load_a_frag(&sA[cur][wm * 32 + sm * 16 + lo][0], hi);
#pragma unroll
    for (int sn = 0; sn < 2; ++sn) {
      v16bf b1 = load_b_frag(&sB1[cur][wn * 32 + sn * 16 + lo][0], hi);
      v16bf b3 = load_b_frag(&sB3[cur][wn * 32 + sn * 16 + lo][0], hi);
#pragma unroll
      for (int sm = 0; sm < 2; ++sm) {
        accU[sm][sn] = __builtin_amdgcn_wmma_f32_16x16x32_bf16(
            false, af[sm], false, b1, (short)0, accU[sm][sn], false, false);
        accV[sm][sn] = __builtin_amdgcn_wmma_f32_16x16x32_bf16(
            false, af[sm], false, b3, (short)0, accV[sm][sn], false, false);
      }
    }
    __syncthreads();                   // all waves done with buf before reuse
  }

#pragma unroll
  for (int sm = 0; sm < 2; ++sm)
#pragma unroll
    for (int sn = 0; sn < 2; ++sn)
#pragma unroll
      for (int v = 0; v < 8; ++v) {
        int row = m0 + wm * 32 + sm * 16 + hi * 8 + v;
        int col = n0 + wn * 32 + sn * 16 + lo;
        float u = accU[sm][sn][v], w = accV[sm][sn][v];
        float h = (u / (1.f + __expf(-u))) * w;  // silu(u) * w
        hb[((size_t)e * T_TOK + row) * FDIM + col] = f2bf(h);
      }
}

// ---------------------------------------------------------------------------
// GEMM2: out[tok] += w(tok,e) * (H @ W2).  Same tiling; 3 async copies per
// thread per k-step; scatter via f32 global atomic add.
// ---------------------------------------------------------------------------
__global__ __launch_bounds__(256) void gemm2_down(
    const unsigned short* __restrict__ hb,   // [E, T, F] bf16
    const unsigned short* __restrict__ w2b,  // [E, D, F] bf16 (N-major)
    float* __restrict__ out,                 // [T, D] f32 (pre-zeroed)
    const int* __restrict__ cnt, const int* __restrict__ idx,
    const float* __restrict__ gwt) {
  int e = blockIdx.z;
  int n0 = blockIdx.x * 128;  // over D
  int m0 = blockIdx.y * 64;
  int ce = cnt[e];
  if (m0 >= ce) return;

  __shared__ __align__(32) unsigned short sA[2][64][32];
  __shared__ __align__(32) unsigned short sB[2][128][32];
  __shared__ int sIdx[64];
  __shared__ float sW[64];

  int tid = threadIdx.x, lane = tid & 31, wave = tid >> 5;
  int wm = wave >> 2, wn = wave & 3;
  int hi = lane >> 4, lo = lane & 15;

  if (tid < 64) {
    int r = m0 + tid;
    sIdx[tid] = (r < ce) ? idx[e * T_TOK + r] : 0;
    sW[tid]   = (r < ce) ? gwt[e * T_TOK + r] : 0.f;
  }
  __syncthreads();

  const unsigned short* hbase = hb + (size_t)e * T_TOK * FDIM;
  int ar = tid >> 2, ach = (tid & 3) * 8;
  int br = tid >> 1, bhf = (tid & 1) * 16;

  auto stage = [&](int buf, int k0) {
    async_cp16(&sA[buf][ar][ach], hbase + (size_t)(m0 + ar) * FDIM + k0 + ach);
    const unsigned short* bs = w2b + ((size_t)e * DMODEL + n0 + br) * FDIM + k0 + bhf;
    async_cp16(&sB[buf][br][bhf],     bs);
    async_cp16(&sB[buf][br][bhf + 8], bs + 8);
  };

  v8f acc[2][2] = {{{0.f}}};

  const int NKB = FDIM / 32;
  stage(0, 0);
  for (int kb = 0; kb < NKB; ++kb) {
    int cur = kb & 1;
    if (kb + 1 < NKB) {
      stage(cur ^ 1, (kb + 1) * 32);
      wait_async_le(3);
    } else {
      wait_async_le(0);
    }
    __syncthreads();

    v16bf af[2];
#pragma unroll
    for (int sm = 0; sm < 2; ++sm)
      af[sm] = load_a_frag(&sA[cur][wm * 32 + sm * 16 + lo][0], hi);
#pragma unroll
    for (int sn = 0; sn < 2; ++sn) {
      v16bf bf = load_b_frag(&sB[cur][wn * 32 + sn * 16 + lo][0], hi);
#pragma unroll
      for (int sm = 0; sm < 2; ++sm)
        acc[sm][sn] = __builtin_amdgcn_wmma_f32_16x16x32_bf16(
            false, af[sm], false, bf, (short)0, acc[sm][sn], false, false);
    }
    __syncthreads();
  }

#pragma unroll
  for (int sm = 0; sm < 2; ++sm)
#pragma unroll
    for (int sn = 0; sn < 2; ++sn)
#pragma unroll
      for (int v = 0; v < 8; ++v) {
        int rl = wm * 32 + sm * 16 + hi * 8 + v;
        if (m0 + rl < ce) {
          int tok = sIdx[rl];
          float val = sW[rl] * acc[sm][sn][v];
          int col = n0 + wn * 32 + sn * 16 + lo;
          unsafeAtomicAdd(&out[(size_t)tok * DMODEL + col], val);
        }
      }
}

// ---------------------------------------------------------------------------
// Launch. d_in: inputs_raw[T,D], gate_w[D,E], w1[E,D,F], w3[E,D,F], w2[E,F,D].
// d_out: results (T*D f32) | l_aux (1) | gate_logits (T*E).
// d_ws layout: cnt/me/ce (1KB) | idx (E*T i32) | gwt (E*T f32) | xb | w1b |
//              w3b | w2b | hb   (~243 MB total)
// ---------------------------------------------------------------------------
extern "C" void kernel_launch(void* const* d_in, const int* in_sizes, int n_in,
                              void* d_out, int out_size, void* d_ws, size_t ws_size,
                              hipStream_t stream) {
  const float* x   = (const float*)d_in[0];
  const float* gw  = (const float*)d_in[1];
  const float* w1  = (const float*)d_in[2];
  const float* w3  = (const float*)d_in[3];
  const float* w2  = (const float*)d_in[4];

  float* out  = (float*)d_out;
  float* laux = out + (size_t)T_TOK * DMODEL;
  float* glog = laux + 1;

  char* ws = (char*)d_ws;
  int*   cnt   = (int*)(ws + 0);
  float* me    = (float*)(ws + 256);
  float* cesum = (float*)(ws + 512);
  int*   idx   = (int*)(ws + 1024);
  float* gwt   = (float*)(ws + 1024 + (size_t)NEXP * T_TOK * 4);
  unsigned short* xb  = (unsigned short*)(ws + 1024 + 2 * (size_t)NEXP * T_TOK * 4);
  unsigned short* w1b = xb  + (size_t)T_TOK * DMODEL;
  unsigned short* w3b = w1b + (size_t)NEXP * DMODEL * FDIM;
  unsigned short* w2b = w3b + (size_t)NEXP * DMODEL * FDIM;
  unsigned short* hb  = w2b + (size_t)NEXP * DMODEL * FDIM;

  hipMemsetAsync(d_out, 0, (size_t)T_TOK * DMODEL * sizeof(float), stream);
  hipMemsetAsync(ws, 0, 1024, stream);  // counters + me/ce sums

  cvt_x<<<(T_TOK * DMODEL + 255) / 256, 256, 0, stream>>>(x, xb, T_TOK * DMODEL);
  tconv<<<dim3(FDIM / 32, DMODEL / 32, NEXP), 256, 0, stream>>>(w1, w1b, DMODEL, FDIM);
  tconv<<<dim3(FDIM / 32, DMODEL / 32, NEXP), 256, 0, stream>>>(w3, w3b, DMODEL, FDIM);
  tconv<<<dim3(DMODEL / 32, FDIM / 32, NEXP), 256, 0, stream>>>(w2, w2b, FDIM, DMODEL);

  gate_kernel<<<T_TOK / 8, 256, 0, stream>>>(x, gw, glog, cnt, me, cesum, idx, gwt);
  laux_kernel<<<1, 1, 0, stream>>>(me, cesum, laux);

  gemm1_swiglu<<<dim3(FDIM / 128, T_TOK / 64, NEXP), 256, 0, stream>>>(
      xb, w1b, w3b, hb, cnt, idx);
  gemm2_down<<<dim3(DMODEL / 128, T_TOK / 64, NEXP), 256, 0, stream>>>(
      hb, w2b, out, cnt, idx, gwt);
}
